// GRU_Cell_40621800686224
// MI455X (gfx1250) — matrix operand order repaired and checked by reading, not compiled, not run on hardware
//
#include <hip/hip_runtime.h>
#include <math.h>

// ---------------------------------------------------------------------------
// GRU cell on MI455X (gfx1250, wave32, WMMA).
// B=16384, IN=H=1024. 6 GEMMs (~206 GFLOP) -> bf16 WMMA 16x16x32.
// Phase 0: convert x/h/weights f32->bf16 (weights: 12MB bf16, L2 resident).
// Phase 1: gates kernel, dual GEMM K=2048 ([x|h]) -> u (f32), hr=h*sigmoid(r) (bf16).
// Phase 2: candidate kernel, GEMM K=2048 ([x|hr]) -> out = (1-u)*h + u*tanh(.).
// ---------------------------------------------------------------------------

#define BATCH 16384
#define INDIM 1024
#define HDIM  1024

typedef __attribute__((ext_vector_type(16))) __bf16 bf16x16;
typedef __attribute__((ext_vector_type(8)))  __bf16 bf16x8;
typedef __attribute__((ext_vector_type(4)))  __bf16 bf16x4;
typedef __attribute__((ext_vector_type(8)))  float  f32x8;
typedef __attribute__((ext_vector_type(4)))  float  f32x4;

// ---------------- f32 -> bf16 conversion (vectorized x4) -------------------
__global__ __launch_bounds__(256) void f32_to_bf16_kernel(
    const float* __restrict__ src, __bf16* __restrict__ dst, int n4) {
  int i = blockIdx.x * blockDim.x + threadIdx.x;
  if (i < n4) {
    f32x4 v = ((const f32x4*)src)[i];
    bf16x4 o;
    o.x = (__bf16)v.x; o.y = (__bf16)v.y; o.z = (__bf16)v.z; o.w = (__bf16)v.w;
    ((bf16x4*)dst)[i] = o;
  }
}

// ---------------- WMMA fragment helpers ------------------------------------
// A fragment 16x32 bf16: lane l holds row m0+(l&15); K chunk
// [16*hi, 16*hi+16)  (hi = l>>4). Two 16B loads, K-major source.
__device__ __forceinline__ bf16x16 load_a_frag(const __bf16* __restrict__ base,
                                               int ld, int m0, int k0, int lane) {
  int row = m0 + (lane & 15);
  int hi  = lane >> 4;
  const __bf16* p = base + (size_t)row * ld + k0 + 16 * hi;
  union { bf16x16 v; bf16x8 h[2]; } u;
  u.h[0] = *(const bf16x8*)(p);
  u.h[1] = *(const bf16x8*)(p + 8);
  return u.v;
}

// B fragment 32x16 bf16: lane l holds column n0+(l&15); K chunk
// [16*hi, 16*hi+16). One contiguous 32B load (weights are [out,in], K-major).
__device__ __forceinline__ bf16x16 load_b_frag(const __bf16* __restrict__ base,
                                               int ld, int n0, int k0, int lane) {
  int col = n0 + (lane & 15);
  int hi  = lane >> 4;
  return *(const bf16x16*)(base + (size_t)col * ld + k0 + 16 * hi);
}

__device__ __forceinline__ f32x8 wmma_bf16(bf16x16 a, bf16x16 b, f32x8 c) {
  return __builtin_amdgcn_wmma_f32_16x16x32_bf16(
      /*neg_a=*/false, a, /*neg_b=*/false, b,
      /*c_mod=*/(short)0, c, /*reuse_a=*/false, /*reuse_b=*/false);
}

__device__ __forceinline__ float sigmoidf_fast(float x) {
  return 1.0f / (1.0f + __expf(-x));
}

// K-loop accumulating a 32x32 tile into TWO gate accumulators (shared A).
__device__ __forceinline__ void mma_loop2(const __bf16* __restrict__ A,
                                          const __bf16* __restrict__ B1,
                                          const __bf16* __restrict__ B2,
                                          int K, int m0, int n0, int lane,
                                          f32x8 acc1[2][2], f32x8 acc2[2][2]) {
  for (int k0 = 0; k0 < K; k0 += 32) {
    bf16x16 a0  = load_a_frag(A,  K, m0,      k0, lane);
    bf16x16 a1  = load_a_frag(A,  K, m0 + 16, k0, lane);
    bf16x16 b10 = load_b_frag(B1, K, n0,      k0, lane);
    bf16x16 b11 = load_b_frag(B1, K, n0 + 16, k0, lane);
    bf16x16 b20 = load_b_frag(B2, K, n0,      k0, lane);
    bf16x16 b21 = load_b_frag(B2, K, n0 + 16, k0, lane);
    acc1[0][0] = wmma_bf16(a0, b10, acc1[0][0]);
    acc1[0][1] = wmma_bf16(a0, b11, acc1[0][1]);
    acc1[1][0] = wmma_bf16(a1, b10, acc1[1][0]);
    acc1[1][1] = wmma_bf16(a1, b11, acc1[1][1]);
    acc2[0][0] = wmma_bf16(a0, b20, acc2[0][0]);
    acc2[0][1] = wmma_bf16(a0, b21, acc2[0][1]);
    acc2[1][0] = wmma_bf16(a1, b20, acc2[1][0]);
    acc2[1][1] = wmma_bf16(a1, b21, acc2[1][1]);
  }
}

// K-loop, single accumulator set.
__device__ __forceinline__ void mma_loop1(const __bf16* __restrict__ A,
                                          const __bf16* __restrict__ B,
                                          int K, int m0, int n0, int lane,
                                          f32x8 acc[2][2]) {
  for (int k0 = 0; k0 < K; k0 += 32) {
    bf16x16 a0 = load_a_frag(A, K, m0,      k0, lane);
    bf16x16 a1 = load_a_frag(A, K, m0 + 16, k0, lane);
    bf16x16 b0 = load_b_frag(B, K, n0,      k0, lane);
    bf16x16 b1 = load_b_frag(B, K, n0 + 16, k0, lane);
    acc[0][0] = wmma_bf16(a0, b0, acc[0][0]);
    acc[0][1] = wmma_bf16(a0, b1, acc[0][1]);
    acc[1][0] = wmma_bf16(a1, b0, acc[1][0]);
    acc[1][1] = wmma_bf16(a1, b1, acc[1][1]);
  }
}

// ---------------- Phase 1: reset/update gates -------------------------------
// r = sigmoid(x@W_r.T + h@U_r.T + bWr + bUr); u likewise.
// Emits: u_ws (f32), hr_ws = bf16(h_prev * r).
__global__ __launch_bounds__(256) void gru_gates_kernel(
    const __bf16* __restrict__ xb,  const __bf16* __restrict__ hb,
    const __bf16* __restrict__ Wrb, const __bf16* __restrict__ Urb,
    const __bf16* __restrict__ Wub, const __bf16* __restrict__ Uub,
    const float* __restrict__ bWr,  const float* __restrict__ bUr,
    const float* __restrict__ bWu,  const float* __restrict__ bUu,
    const float* __restrict__ h_prev,
    float* __restrict__ u_ws, __bf16* __restrict__ hr_ws) {
  const int lane = threadIdx.x & 31;
  const int wave = threadIdx.x >> 5;          // 8 waves / workgroup
  const int m0 = blockIdx.x * 64  + (wave >> 2) * 32;   // batch tile
  const int n0 = blockIdx.y * 128 + (wave & 3) * 32;    // hidden tile

  f32x8 accR[2][2] = {}; f32x8 accU[2][2] = {};
  mma_loop2(xb, Wrb, Wub, INDIM, m0, n0, lane, accR, accU);  // x phase
  mma_loop2(hb, Urb, Uub, HDIM,  m0, n0, lane, accR, accU);  // h phase

  const int hi = lane >> 4, nn = lane & 15;
  #pragma unroll
  for (int j = 0; j < 2; ++j) {
    const int col = n0 + 16 * j + nn;
    const float biasR = bWr[col] + bUr[col];
    const float biasU = bWu[col] + bUu[col];
    #pragma unroll
    for (int i = 0; i < 2; ++i) {
      #pragma unroll
      for (int v = 0; v < 8; ++v) {
        const int row = m0 + 16 * i + 8 * hi + v;   // C/D layout: M = v + 8*hi
        const size_t idx = (size_t)row * HDIM + col;
        const float r = sigmoidf_fast(accR[i][j][v] + biasR);
        const float u = sigmoidf_fast(accU[i][j][v] + biasU);
        u_ws[idx]  = u;
        hr_ws[idx] = (__bf16)(h_prev[idx] * r);
      }
    }
  }
}

// ---------------- Phase 2: candidate + blend --------------------------------
// c = tanh(x@W.T + hr@U.T + bW + bU); out = (1-u)*h_prev + u*c.
__global__ __launch_bounds__(256) void gru_cand_kernel(
    const __bf16* __restrict__ xb, const __bf16* __restrict__ hrb,
    const __bf16* __restrict__ Wb, const __bf16* __restrict__ Ub,
    const float* __restrict__ bW,  const float* __restrict__ bU,
    const float* __restrict__ u_ws, const float* __restrict__ h_prev,
    float* __restrict__ out) {
  const int lane = threadIdx.x & 31;
  const int wave = threadIdx.x >> 5;
  const int m0 = blockIdx.x * 64  + (wave >> 2) * 32;
  const int n0 = blockIdx.y * 128 + (wave & 3) * 32;

  f32x8 acc[2][2] = {};
  mma_loop1(xb,  Wb, INDIM, m0, n0, lane, acc);   // x phase
  mma_loop1(hrb, Ub, HDIM,  m0, n0, lane, acc);   // (h*r) phase

  const int hi = lane >> 4, nn = lane & 15;
  #pragma unroll
  for (int j = 0; j < 2; ++j) {
    const int col = n0 + 16 * j + nn;
    const float bias = bW[col] + bU[col];
    #pragma unroll
    for (int i = 0; i < 2; ++i) {
      #pragma unroll
      for (int v = 0; v < 8; ++v) {
        const int row = m0 + 16 * i + 8 * hi + v;
        const size_t idx = (size_t)row * HDIM + col;
        const float c = tanhf(acc[i][j][v] + bias);
        const float u = u_ws[idx];
        out[idx] = (1.0f - u) * h_prev[idx] + u * c;
      }
    }
  }
}

// ---------------- Host-side orchestration -----------------------------------
extern "C" void kernel_launch(void* const* d_in, const int* in_sizes, int n_in,
                              void* d_out, int out_size, void* d_ws, size_t ws_size,
                              hipStream_t stream) {
  const float* x_t    = (const float*)d_in[0];
  const float* h_prev = (const float*)d_in[1];
  const float* W_r    = (const float*)d_in[2];
  const float* b_Wr   = (const float*)d_in[3];
  const float* U_r    = (const float*)d_in[4];
  const float* b_Ur   = (const float*)d_in[5];
  const float* W_u    = (const float*)d_in[6];
  const float* b_Wu   = (const float*)d_in[7];
  const float* U_u    = (const float*)d_in[8];
  const float* b_Uu   = (const float*)d_in[9];
  const float* W      = (const float*)d_in[10];
  const float* b_W    = (const float*)d_in[11];
  const float* U      = (const float*)d_in[12];
  const float* b_U    = (const float*)d_in[13];
  float* out = (float*)d_out;

  // Workspace carve-up (~172 MB total).
  char* ws = (char*)d_ws;
  const size_t xh_b  = (size_t)BATCH * INDIM * sizeof(__bf16);   // 32 MB
  const size_t w_b   = (size_t)HDIM * INDIM * sizeof(__bf16);    //  2 MB
  __bf16* xb  = (__bf16*)(ws);
  __bf16* hb  = (__bf16*)(ws + xh_b);
  __bf16* Wrb = (__bf16*)(ws + 2 * xh_b);
  __bf16* Urb = (__bf16*)(ws + 2 * xh_b + 1 * w_b);
  __bf16* Wub = (__bf16*)(ws + 2 * xh_b + 2 * w_b);
  __bf16* Uub = (__bf16*)(ws + 2 * xh_b + 3 * w_b);
  __bf16* Wb  = (__bf16*)(ws + 2 * xh_b + 4 * w_b);
  __bf16* Ub  = (__bf16*)(ws + 2 * xh_b + 5 * w_b);
  float*  u_ws  = (float*) (ws + 2 * xh_b + 6 * w_b);            // 64 MB
  __bf16* hr_ws = (__bf16*)(ws + 2 * xh_b + 6 * w_b +
                            (size_t)BATCH * HDIM * sizeof(float));

  // Phase 0: downconvert activations + weights to bf16.
  const int n4_xh = (BATCH * INDIM) / 4, n4_w = (HDIM * INDIM) / 4;
  f32_to_bf16_kernel<<<n4_xh / 256, 256, 0, stream>>>(x_t,    xb,  n4_xh);
  f32_to_bf16_kernel<<<n4_xh / 256, 256, 0, stream>>>(h_prev, hb,  n4_xh);
  f32_to_bf16_kernel<<<n4_w  / 256, 256, 0, stream>>>(W_r,    Wrb, n4_w);
  f32_to_bf16_kernel<<<n4_w  / 256, 256, 0, stream>>>(U_r,    Urb, n4_w);
  f32_to_bf16_kernel<<<n4_w  / 256, 256, 0, stream>>>(W_u,    Wub, n4_w);
  f32_to_bf16_kernel<<<n4_w  / 256, 256, 0, stream>>>(U_u,    Uub, n4_w);
  f32_to_bf16_kernel<<<n4_w  / 256, 256, 0, stream>>>(W,      Wb,  n4_w);
  f32_to_bf16_kernel<<<n4_w  / 256, 256, 0, stream>>>(U,      Ub,  n4_w);

  // Phase 1 + 2: 64x128 workgroup tiles, 8 waves, 32x32 per wave.
  dim3 grid(BATCH / 64, HDIM / 128), block(256);
  gru_gates_kernel<<<grid, block, 0, stream>>>(xb, hb, Wrb, Urb, Wub, Uub,
                                               b_Wr, b_Ur, b_Wu, b_Uu,
                                               h_prev, u_ws, hr_ws);
  gru_cand_kernel<<<grid, block, 0, stream>>>(xb, hr_ws, Wb, Ub, b_W, b_U,
                                              u_ws, h_prev, out);
}
